// VectorQuantizer_88811333747152
// MI455X (gfx1250) — matrix operand order repaired and checked by reading, not compiled, not run on hardware
//
#include <hip/hip_runtime.h>
#include <hip/hip_bf16.h>
#include <math.h>

// ---------------------------------------------------------------------------
// VQ (dual codebook) for MI455X / gfx1250, wave32.
//   rows are unit-norm =>  d_text = 2 - 2*<zt,et>,  d_graph = 2 - 2*<zg,eg>
//   Heavy work: 8192x8192x1024 bf16 GEMM -> v_wmma_f32_16x16x32_bf16,
//   B tiles staged LDS via GLOBAL_LOAD_ASYNC_TO_LDS_B128 (double buffered).
// ---------------------------------------------------------------------------

#define B_ROWS   8192
#define N_EMB    8192
#define D_FULL   1024
#define D_HALF   512
#define INV_T    100.0f   // 1/TEMPERATURE

typedef __attribute__((ext_vector_type(16))) __bf16 v16bf;
typedef __attribute__((ext_vector_type(8)))  float  v8f;
typedef __attribute__((ext_vector_type(4)))  unsigned int u32x4;
typedef __attribute__((vector_size(16)))     int    v4i;

// ---- async global->LDS copy (CDNA5 ASYNCcnt path), with safe fallbacks -----
#if defined(__HIP_DEVICE_COMPILE__) && __has_builtin(__builtin_amdgcn_global_load_async_to_lds_b128)
#define USE_ASYNC_LDS 1
#else
#define USE_ASYNC_LDS 0
#endif

__device__ __forceinline__ void copy16_g2l(__bf16* ldst, const __bf16* gsrc) {
#if USE_ASYNC_LDS
  __builtin_amdgcn_global_load_async_to_lds_b128((v4i*)gsrc, (v4i*)ldst, 0, 0);
#else
  *(u32x4*)ldst = *(const u32x4*)gsrc;
#endif
}
__device__ __forceinline__ void wait_async_copies() {
#if USE_ASYNC_LDS
#if __has_builtin(__builtin_amdgcn_s_wait_asynccnt)
  __builtin_amdgcn_s_wait_asynccnt(0);
#else
  asm volatile("s_wait_asynccnt 0x0" ::: "memory");
#endif
#endif
}

// ---- WMMA fragment loaders (contiguous k order, same for A and B, wave32) ----------
// A 16x32 bf16: lanes 0-15 = rows M0-15 holding K {k0..k0+15} contiguous;
//               lanes 16-31 = same rows holding K {k0+16..k0+31}.
__device__ __forceinline__ v16bf load_frag_a(const __bf16* rowbase, int stride,
                                             int k0, int lane) {
  const __bf16* p = rowbase + (size_t)(lane & 15) * stride + k0 + ((lane & 16) ? 16 : 0);
  union { u32x4 u[2]; v16bf v; } r;
  r.u[0] = *(const u32x4*)(p);
  r.u[1] = *(const u32x4*)(p + 8);
  return r.v;
}
// B 32x16 bf16 (B[k][n] = E[n][k], E row-major): lanes 0-15 = col n holding
// K {k0..k0+15} contiguous; lanes 16-31 hold K {k0+16..k0+31}.
__device__ __forceinline__ v16bf load_frag_b(const __bf16* rowbase, int stride,
                                             int k0, int lane) {
  const __bf16* p = rowbase + (size_t)(lane & 15) * stride + k0 + ((lane & 16) ? 16 : 0);
  union { u32x4 u[2]; v16bf v; } r;
  r.u[0] = *(const u32x4*)(p);
  r.u[1] = *(const u32x4*)(p + 8);
  return r.v;
}
// Same as load_frag_b but from an LDS-staged tile with padded row stride 72
// (72 bf16 = 36 dwords; gcd(36,64)=4 -> conflict-free b128 fragment reads).
#define BROW_PAD   72
#define BBUF_ELEMS (128 * BROW_PAD)
__device__ __forceinline__ v16bf load_frag_b_lds(const __bf16* buf, int rowbase,
                                                 int koff, int lane) {
  const __bf16* p = buf + (rowbase + (lane & 15)) * BROW_PAD + koff + ((lane & 16) ? 16 : 0);
  union { u32x4 u[2]; v16bf v; } r;
  r.u[0] = *(const u32x4*)(p);
  r.u[1] = *(const u32x4*)(p + 8);
  return r.v;
}
__device__ __forceinline__ v8f wmma_bf16(v16bf a, v16bf b, v8f c) {
  return __builtin_amdgcn_wmma_f32_16x16x32_bf16(false, a, false, b,
                                                 (short)0, c, false, false);
}

// Stage one B chunk: rows [n0, n0+128), k = [kc*64, kc*64+64) -> LDS buffer.
__device__ __forceinline__ void stage_b(__bf16* dst, const __bf16* Eb,
                                        int n0, int kc, int tid) {
  #pragma unroll
  for (int it = 0; it < 4; ++it) {          // 1024 x 16B segments / 256 thr
    const int s    = tid + it * 256;
    const int row  = s >> 3;
    const int off8 = (s & 7) * 8;           // element offset within 64-wide k
    copy16_g2l(dst + row * BROW_PAD + off8,
               Eb + (size_t)(n0 + row) * D_FULL + kc * 64 + off8);
  }
}

// ---- init: zero accumulators in workspace ----------------------------------
__global__ __launch_bounds__(256) void vq_init(float* avgp, float* scal) {
  int t = blockIdx.x * 256 + threadIdx.x;
  if (t < N_EMB) avgp[t] = 0.0f;
  if (t < 8) scal[t] = 0.0f;
}

// ---- row L2-normalize, emit fp32 + bf16 copies -----------------------------
__global__ __launch_bounds__(256)
void norm_rows(const float* __restrict__ src, int src_stride, int src_off,
               float* __restrict__ dstF, __bf16* __restrict__ dstB,
               int dst_stride, int dst_off, int ncols) {
  __shared__ float buf[256];
  const float* s = src + (size_t)blockIdx.x * src_stride + src_off;
  float ss = 0.0f;
  for (int c = threadIdx.x; c < ncols; c += 256) { float v = s[c]; ss += v * v; }
  buf[threadIdx.x] = ss;
  __syncthreads();
  for (int st = 128; st > 0; st >>= 1) {
    if ((int)threadIdx.x < st) buf[threadIdx.x] += buf[threadIdx.x + st];
    __syncthreads();
  }
  float inv = 1.0f / sqrtf(buf[0]);
  float*  dF = dstF + (size_t)blockIdx.x * dst_stride + dst_off;
  __bf16* dB = dstB + (size_t)blockIdx.x * dst_stride + dst_off;
  for (int c = threadIdx.x; c < ncols; c += 256) {
    float v = s[c] * inv;
    dF[c] = v;
    dB[c] = (__bf16)v;
  }
}

// ---- pass 1: GEMM + fused row reductions -----------------------------------
// Block tile: 64 rows x 128 cols.  8 waves: wave w owns m-subtile (w>>1) and
// n-columns [(w&1)*64, (w&1)*64+64) as 4 subtiles of 16.
// B tiles: async-staged into double-buffered LDS (ASYNCcnt); A direct global.
__global__ __launch_bounds__(256)
void vq_gemm_reduce(const __bf16* __restrict__ Zb, const __bf16* __restrict__ Eb,
                    float* __restrict__ row_m, float* __restrict__ row_lZ,
                    int* __restrict__ idx_out, float* __restrict__ idx_out_f,
                    float* __restrict__ scal) {
  // 64KB region, time-multiplexed: (a) B double-buffer (2 x 18KB, bf16),
  // (b) S_text/S_graph 64x128 f32 tiles. Separated by barriers.
  __shared__ __align__(16) float smem[64 * 128 * 2];
  __shared__ float p_min[256], p_dt2[256], p_dg2[256], p_m[256], p_Ze[256], p_Sa[256];
  __shared__ int   p_idx[256];
  __shared__ float r_min[64], r_dt2[64], r_dg2[64], r_m[64], r_Ze[64], r_Sa[64];
  __shared__ int   r_idx[64];

  float*  Stf = smem;                 // [64][128]
  float*  Sgf = smem + 64 * 128;      // [64][128]
  __bf16* bst = (__bf16*)smem;        // staging buffers 0/1

  const int tid   = threadIdx.x;
  const int lane  = tid & 31, wave = tid >> 5;
  const int m_sub = wave >> 1;             // 0..3
  const int nbase = (wave & 1) * 64;       // 0 or 64
  const int cl = lane & 15, ch = lane >> 4;

  const __bf16* Arow = Zb + (size_t)(blockIdx.x * 64 + m_sub * 16) * D_FULL;

  if (tid < 64) {
    r_min[tid] = 3.4e38f; r_idx[tid] = 0;
    r_dt2[tid] = 0.0f; r_dg2[tid] = 0.0f;
    r_m[tid] = -3.4e38f; r_Ze[tid] = 0.0f; r_Sa[tid] = 0.0f;
  }
  __syncthreads();

  for (int n0 = 0; n0 < N_EMB; n0 += 128) {
    v8f accT[4] = {}, accG[4] = {};
    int cur = 0;
    stage_b(bst, Eb, n0, 0, tid);                 // prefetch chunk 0

    // ---- text half: k chunks 0..7 (k = 0..511) ----
    #pragma unroll 2
    for (int kc = 0; kc < 8; ++kc) {
      wait_async_copies();
      __syncthreads();                            // buf[cur] ready for all
      stage_b(bst + (cur ^ 1) * BBUF_ELEMS, Eb, n0, kc + 1, tid);
      const __bf16* bcur = bst + cur * BBUF_ELEMS;
      #pragma unroll
      for (int ks = 0; ks < 2; ++ks) {
        v16bf a = load_frag_a(Arow, D_FULL, kc * 64 + ks * 32, lane);
        v16bf b[4];
        #pragma unroll
        for (int ns = 0; ns < 4; ++ns)
          b[ns] = load_frag_b_lds(bcur, nbase + ns * 16, ks * 32, lane);
        #pragma unroll
        for (int ns = 0; ns < 4; ++ns)
          accT[ns] = wmma_bf16(a, b[ns], accT[ns]);
      }
      __syncthreads();                            // done reading buf[cur]
      cur ^= 1;
    }
    // ---- graph half: k chunks 8..15 (k = 512..1023) ----
    #pragma unroll 2
    for (int kc = 8; kc < 16; ++kc) {
      wait_async_copies();
      __syncthreads();
      if (kc + 1 < 16)
        stage_b(bst + (cur ^ 1) * BBUF_ELEMS, Eb, n0, kc + 1, tid);
      const __bf16* bcur = bst + cur * BBUF_ELEMS;
      #pragma unroll
      for (int ks = 0; ks < 2; ++ks) {
        v16bf a = load_frag_a(Arow, D_FULL, kc * 64 + ks * 32, lane);
        v16bf b[4];
        #pragma unroll
        for (int ns = 0; ns < 4; ++ns)
          b[ns] = load_frag_b_lds(bcur, nbase + ns * 16, ks * 32, lane);
        #pragma unroll
        for (int ns = 0; ns < 4; ++ns)
          accG[ns] = wmma_bf16(a, b[ns], accG[ns]);
      }
      __syncthreads();
      cur ^= 1;
    }

    // ---- spill S tiles to LDS (C layout: VGPR r, lanes0-15 => M=r; 16-31 => M=r+8)
    const int mrowb = m_sub * 16 + ch * 8;
    #pragma unroll
    for (int ns = 0; ns < 4; ++ns) {
      const int ncol = nbase + ns * 16 + cl;
      #pragma unroll
      for (int r = 0; r < 8; ++r) {
        Stf[(mrowb + r) * 128 + ncol] = accT[ns][r];
        Sgf[(mrowb + r) * 128 + ncol] = accG[ns][r];
      }
    }
    __syncthreads();

    // ---- per-thread partials: 4 threads per row, 32 cols each
    {
      const int row = tid >> 2, q = tid & 3;
      float dmin = 3.4e38f, dt2 = 0.0f, dg2 = 0.0f;
      float mm = -3.4e38f, Ze = 0.0f, Sa = 0.0f;
      int imin = 0;
      #pragma unroll 8
      for (int c = 0; c < 32; ++c) {
        const int col = q * 32 + c;
        const float st = Stf[row * 128 + col], sg = Sgf[row * 128 + col];
        const float dt = 2.0f - 2.0f * st, dg = 2.0f - 2.0f * sg;
        const float d  = dt + dg;
        dt2 += dt * dt; dg2 += dg * dg;
        if (d < dmin) { dmin = d; imin = n0 + col; }
        const float aa = -d * INV_T;              // a = -d / T
        if (aa > mm) { float s = __expf(mm - aa); Ze *= s; Sa *= s; mm = aa; }
        const float e = __expf(aa - mm);
        Ze += e; Sa += aa * e;
      }
      p_min[tid] = dmin; p_idx[tid] = imin;
      p_dt2[tid] = dt2;  p_dg2[tid] = dg2;
      p_m[tid] = mm; p_Ze[tid] = Ze; p_Sa[tid] = Sa;
    }
    __syncthreads();

    if (tid < 64) {  // merge 4 partials into running row stats
      float dminr = r_min[tid]; int iminr = r_idx[tid];
      float dt2r = r_dt2[tid], dg2r = r_dg2[tid];
      float mr = r_m[tid], Zer = r_Ze[tid], Sar = r_Sa[tid];
      #pragma unroll
      for (int qq = 0; qq < 4; ++qq) {
        const int j = tid * 4 + qq;
        if (p_min[j] < dminr) { dminr = p_min[j]; iminr = p_idx[j]; }
        dt2r += p_dt2[j]; dg2r += p_dg2[j];
        const float mo = p_m[j], Zo = p_Ze[j], So = p_Sa[j];
        if (mo > mr) { float s = __expf(mr - mo); Zer *= s; Sar *= s; mr = mo; }
        const float s2 = __expf(mo - mr);
        Zer += Zo * s2; Sar += So * s2;
      }
      r_min[tid] = dminr; r_idx[tid] = iminr;
      r_dt2[tid] = dt2r;  r_dg2[tid] = dg2r;
      r_m[tid] = mr; r_Ze[tid] = Zer; r_Sa[tid] = Sar;
    }
    __syncthreads();                              // S/B region free again
  }

  if (tid < 64) {
    const int g = blockIdx.x * 64 + tid;
    idx_out[g]   = r_idx[tid];
    idx_out_f[g] = (float)r_idx[tid];
    const float lz = __logf(r_Ze[tid]);
    row_m[g]  = r_m[tid];
    row_lZ[g] = lz;
    // sum_j p*logp = E_p[a] - lse(a); sample entropy contribution = -that
    const float se = -(r_Sa[tid] / r_Ze[tid] - r_m[tid] - lz);
    atomicAdd(&scal[0], r_dt2[tid]);
    atomicAdd(&scal[1], r_dg2[tid]);
    atomicAdd(&scal[2], se);
  }
}

// ---- pass 2: recompute S (full K=1024 chain) -> avg_probs column sums ------
// a = -d/T = 200*S - 400 ;  p_ij = exp(a_ij - m_i - logZ_i)
__global__ __launch_bounds__(256)
void vq_avg_probs(const __bf16* __restrict__ Zb, const __bf16* __restrict__ Eb,
                  const float* __restrict__ row_m, const float* __restrict__ row_lZ,
                  float* __restrict__ avgp) {
  __shared__ __align__(16) float P[64 * 128];
  __shared__ float csum[2][128];

  const int tid   = threadIdx.x;
  const int lane  = tid & 31, wave = tid >> 5;
  const int m_sub = wave >> 1;
  const int nbase = (wave & 1) * 64;
  const int cl = lane & 15, ch = lane >> 4;
  const int n0 = blockIdx.y * 128;

  const __bf16* Arow = Zb + (size_t)(blockIdx.x * 64 + m_sub * 16) * D_FULL;

  v8f acc[4] = {};
  #pragma unroll 2
  for (int ks = 0; ks < 32; ++ks) {
    const int k0 = ks * 32;
    v16bf a = load_frag_a(Arow, D_FULL, k0, lane);
    v16bf b[4];
    #pragma unroll
    for (int ns = 0; ns < 4; ++ns) {
      const __bf16* Brow = Eb + (size_t)(n0 + nbase + ns * 16) * D_FULL;
      b[ns] = load_frag_b(Brow, D_FULL, k0, lane);
    }
    #pragma unroll
    for (int ns = 0; ns < 4; ++ns)
      acc[ns] = wmma_bf16(a, b[ns], acc[ns]);
  }

  const int mrowb = m_sub * 16 + ch * 8;
  #pragma unroll
  for (int r = 0; r < 8; ++r) {
    const int mrow = mrowb + r;
    const float shift = row_m[blockIdx.x * 64 + mrow] + row_lZ[blockIdx.x * 64 + mrow];
    #pragma unroll
    for (int ns = 0; ns < 4; ++ns) {
      P[mrow * 128 + nbase + ns * 16 + cl] =
          __expf(200.0f * acc[ns][r] - 400.0f - shift);
    }
  }
  __syncthreads();

  const int col = tid & 127, half = tid >> 7;
  float s = 0.0f;
  #pragma unroll 8
  for (int r = 0; r < 32; ++r) s += P[(half * 32 + r) * 128 + col];
  csum[half][col] = s;
  __syncthreads();
  if (tid < 128) atomicAdd(&avgp[n0 + tid], csum[0][tid] + csum[1][tid]);
}

// ---- gather z_q outputs + vq/commit squared-diff sum -----------------------
__global__ __launch_bounds__(256)
void vq_gather(const float* __restrict__ ef, const float* __restrict__ zn,
               const int* __restrict__ idxw,
               float* __restrict__ out_zq, float* __restrict__ out_text,
               float* __restrict__ out_graph, float* __restrict__ scal) {
  __shared__ float buf[256];
  const int i   = blockIdx.x;
  const int idx = idxw[i];
  const float* e = ef + (size_t)idx * D_FULL;
  const float* z = zn + (size_t)i * D_FULL;
  float s = 0.0f;
  for (int c = threadIdx.x; c < D_FULL; c += 256) {
    const float v = e[c];
    const float diff = v - z[c];
    s += diff * diff;
    out_zq[(size_t)i * D_FULL + c] = v;     // z_q_out == z_q numerically
    if (c < D_HALF) out_text[(size_t)i * D_HALF + c] = v;
    else            out_graph[(size_t)i * D_HALF + (c - D_HALF)] = v;
  }
  buf[threadIdx.x] = s;
  __syncthreads();
  for (int st = 128; st > 0; st >>= 1) {
    if ((int)threadIdx.x < st) buf[threadIdx.x] += buf[threadIdx.x + st];
    __syncthreads();
  }
  if (threadIdx.x == 0) atomicAdd(&scal[3], buf[0]);
}

// ---- finalize scalar outputs -----------------------------------------------
__global__ __launch_bounds__(256)
void vq_finalize(const float* __restrict__ avgp, const float* __restrict__ scal,
                 float* __restrict__ out_scalars) {
  __shared__ float buf[256];
  float s = 0.0f;
  for (int j = threadIdx.x; j < N_EMB; j += 256) {
    const float ap = avgp[j] * (1.0f / (float)B_ROWS);
    s += ap * __logf(ap + 1e-5f);
  }
  buf[threadIdx.x] = s;
  __syncthreads();
  for (int st = 128; st > 0; st >>= 1) {
    if ((int)threadIdx.x < st) buf[threadIdx.x] += buf[threadIdx.x + st];
    __syncthreads();
  }
  if (threadIdx.x == 0) {
    const float avg_entropy    = -buf[0];
    const float sample_entropy = scal[2] * (1.0f / (float)B_ROWS);
    const float entropy_loss   = 0.1f * (sample_entropy - avg_entropy);
    const float vq = scal[3] * (1.0f / ((float)B_ROWS * (float)D_FULL));
    out_scalars[0] = vq;           // vq_loss
    out_scalars[1] = 0.25f * vq;   // commit_loss (BETA)
    out_scalars[2] = entropy_loss;
    out_scalars[3] = scal[0] * (1.0f / (float)B_ROWS);  // text_d_norm
    out_scalars[4] = scal[1] * (1.0f / (float)B_ROWS);  // graph_d_norm
  }
}

// ---------------------------------------------------------------------------
extern "C" void kernel_launch(void* const* d_in, const int* in_sizes, int n_in,
                              void* d_out, int out_size, void* d_ws, size_t ws_size,
                              hipStream_t stream) {
  const float* z      = (const float*)d_in[0];  // (8192,1024)
  const float* emb_t  = (const float*)d_in[1];  // (8192,512)
  const float* emb_g  = (const float*)d_in[2];  // (8192,512)
  float* out = (float*)d_out;

  // workspace layout (~101 MB)
  float*  zn   = (float*)d_ws;                         // 8192*1024 f32
  float*  ef   = zn + (size_t)B_ROWS * D_FULL;         // 8192*1024 f32
  __bf16* znb  = (__bf16*)(ef + (size_t)B_ROWS * D_FULL);
  __bf16* efb  = znb + (size_t)B_ROWS * D_FULL;
  float*  rowm = (float*)(efb + (size_t)B_ROWS * D_FULL);
  float*  rowz = rowm + B_ROWS;
  float*  avgp = rowz + B_ROWS;
  int*    idxw = (int*)(avgp + N_EMB);
  float*  scal = (float*)(idxw + B_ROWS);

  // output layout (flat concat, return order)
  float* out_zq      = out;                                   // 8192*1024
  float* out_scalars = out + (size_t)B_ROWS * D_FULL;         // 5 scalars
  float* out_text    = out_scalars + 5;                       // 8192*512
  float* out_graph   = out_text + (size_t)B_ROWS * D_HALF;    // 8192*512
  float* out_idx     = out_graph + (size_t)B_ROWS * D_HALF;   // 8192

  vq_init<<<(N_EMB + 255) / 256, 256, 0, stream>>>(avgp, scal);

  norm_rows<<<B_ROWS, 256, 0, stream>>>(z,     D_FULL, 0,      zn, znb, D_FULL, 0,      D_HALF);
  norm_rows<<<B_ROWS, 256, 0, stream>>>(z,     D_FULL, D_HALF, zn, znb, D_FULL, D_HALF, D_HALF);
  norm_rows<<<N_EMB,  256, 0, stream>>>(emb_t, D_HALF, 0,      ef, efb, D_FULL, 0,      D_HALF);
  norm_rows<<<N_EMB,  256, 0, stream>>>(emb_g, D_HALF, 0,      ef, efb, D_FULL, D_HALF, D_HALF);

  vq_gemm_reduce<<<B_ROWS / 64, 256, 0, stream>>>(znb, efb, rowm, rowz, idxw,
                                                  out_idx, scal);

  dim3 g2(B_ROWS / 64, N_EMB / 128);
  vq_avg_probs<<<g2, 256, 0, stream>>>(znb, efb, rowm, rowz, avgp);

  vq_gather<<<B_ROWS, 256, 0, stream>>>(ef, zn, idxw, out_zq, out_text,
                                        out_graph, scal);

  vq_finalize<<<1, 256, 0, stream>>>(avgp, scal, out_scalars);
}
